// LESA_523986010128
// MI455X (gfx1250) — matrix operand order repaired and checked by reading, not compiled, not run on hardware
//
#include <hip/hip_runtime.h>

// ---------------- constants ----------------
#define NB    4
#define CIN   512
#define HW    784
#define G     8
#define QK    32
#define VP    64
#define PE    128
#define BR    512
#define OC2   1024
#define RELW  1567   // 2*HW-1

typedef __attribute__((ext_vector_type(16))) __bf16 v16bf;
typedef __attribute__((ext_vector_type(8)))  float  v8f;
typedef __attribute__((ext_vector_type(4)))  unsigned tdm_u4;
typedef __attribute__((ext_vector_type(8)))  int      tdm_i8;
typedef __attribute__((ext_vector_type(4)))  int      tdm_i4;

union Frag {
    unsigned        u[8];
    unsigned short  s[16];
    v16bf           v;
};

__device__ __forceinline__ unsigned short f2bf(float f) {
    union { float f; unsigned u; } x; x.f = f;
    unsigned r = x.u + 0x7FFFu + ((x.u >> 16) & 1u);   // round-nearest-even
    return (unsigned short)(r >> 16);
}
__device__ __forceinline__ float bf2f(unsigned short h) {
    union { unsigned u; float f; } x; x.u = ((unsigned)h) << 16;
    return x.f;
}
__device__ __forceinline__ v8f wmma_bf16(v16bf a, v16bf b, v8f c) {
    return __builtin_amdgcn_wmma_f32_16x16x32_bf16(false, a, false, b, (short)0, c, false, false);
}
__device__ __forceinline__ int iclamp(int v, int lo, int hi) {
    return v < lo ? lo : (v > hi ? hi : v);
}

// ---- Tensor Data Mover: 2D tile (bf16/2-byte elements) global -> LDS ----
// D# per CDNA5 ISA ch.8: group0 {count=1, lds_addr, global_addr[56:0], type=2},
// group1 {data_size=1(2B), tensor_dim0/1, tile_dim0/1, tensor_dim0_stride}.
// Tiles are fully in-bounds, so tensor dims == tile dims.
// This toolchain exposes the 6-arg builtin: (u32x4, i32x8, i32x4, i32x4, i32x8, cpol).
__device__ __forceinline__ void tdm_load_2d_b16(unsigned lds_addr, const void* gaddr,
                                                unsigned tile_d0, unsigned tile_d1,
                                                unsigned long long stride0_elems) {
    unsigned long long ga = (unsigned long long)(size_t)gaddr;
    tdm_u4 g0;
    g0[0] = 1u;                                         // count=1, user descriptor
    g0[1] = lds_addr;                                   // LDS byte address
    g0[2] = (unsigned)(ga & 0xFFFFFFFFu);               // global_addr[31:0]
    g0[3] = (unsigned)((ga >> 32) & 0x01FFFFFFu)        // global_addr[56:32]
          | (2u << 30);                                 // type=2 ("image")
    tdm_i8 g1;
    g1[0] = 0x00010000;                                 // wg_mask=0, data_size=1 (2 bytes)
    g1[1] = (int)((tile_d0 & 0xFFFFu) << 16);           // tensor_dim0[15:0] @ bits63:48
    g1[2] = (int)(((tile_d0 >> 16) & 0xFFFFu)           // tensor_dim0[31:16]
          | ((tile_d1 & 0xFFFFu) << 16));               // tensor_dim1[15:0]
    g1[3] = (int)(((tile_d1 >> 16) & 0xFFFFu)           // tensor_dim1[31:16]
          | ((tile_d0 & 0xFFFFu) << 16));               // tile_dim0
    g1[4] = (int)(tile_d1 & 0xFFFFu);                   // tile_dim1 (tile_dim2 = 0)
    g1[5] = (int)(stride0_elems & 0xFFFFFFFFull);       // tensor_dim0_stride[31:0]
    g1[6] = (int)((stride0_elems >> 32) & 0xFFFFull);   // stride[47:32] (dim1_stride=0)
    g1[7] = 0;
    tdm_i4 z4 = {0, 0, 0, 0};
    tdm_i8 z8 = {0, 0, 0, 0, 0, 0, 0, 0};
    __builtin_amdgcn_tensor_load_to_lds(g0, g1, z4, z4, z8, 0);
}

// ---------------- fp32 -> bf16 pack ----------------
__global__ __launch_bounds__(256) void cvt_kernel(const float* __restrict__ in,
                                                  unsigned short* __restrict__ out, int n) {
    int e = blockIdx.x * 256 + threadIdx.x;
    if (e < n) out[e] = f2bf(in[e]);
}

// ---------------- grouped 3x3 conv (VALU), writes bf16 ----------------
__global__ __launch_bounds__(256) void conv3_kernel(const float* __restrict__ x,
                                                    const float* __restrict__ W3,
                                                    unsigned short* __restrict__ ubf) {
    int e = blockIdx.x * 256 + threadIdx.x;
    if (e >= NB * BR * HW) return;
    int b   = e / (BR * HW);
    int rem = e % (BR * HW);
    int oc  = rem / HW;
    int p   = rem % HW;
    int y = p / 28, xx = p % 28;
    const float* xb = x + ((size_t)b * CIN + (size_t)(oc >> 6) * 64) * HW;
    const float* wp = W3 + (size_t)oc * 576;   // 64*9
    float acc = 0.f;
    for (int ic = 0; ic < 64; ++ic) {
        const float* xc = xb + (size_t)ic * HW;
        const float* wc = wp + ic * 9;
        #pragma unroll
        for (int ky = 0; ky < 3; ++ky) {
            int yy = y + ky - 1;
            if ((unsigned)yy < 28u) {
                #pragma unroll
                for (int kx = 0; kx < 3; ++kx) {
                    int xc2 = xx + kx - 1;
                    if ((unsigned)xc2 < 28u)
                        acc = fmaf(xc[yy * 28 + xc2], wc[ky * 3 + kx], acc);
                }
            }
        }
    }
    ubf[e] = f2bf(acc);
}

// ---------------- generic bf16 WMMA GEMM with TDM tile staging ----------------
// A: M x K (row major, bf16)  B: K x 784 per batch (bf16)  out: M x 784 per batch
// MODE 0: f32 out, 1: bf16 out, 2: bf16 relu out
template <int MODE>
__global__ __launch_bounds__(256) void gemm_bf16_kernel(const unsigned short* __restrict__ A,
                                                        const unsigned short* __restrict__ B,
                                                        const float* __restrict__ gamma,
                                                        const float* __restrict__ beta,
                                                        void* __restrict__ outp,
                                                        int M, int K) {
    __shared__ unsigned At[128][16];  // 128 rows x 32 bf16 (as u32 pairs)
    __shared__ unsigned Bt[32][8];    // 32 k x 16 n
    const int tid  = threadIdx.x;
    const int lane = tid & 31, w = tid >> 5;
    const int m0 = blockIdx.x * 128;
    const int n0 = blockIdx.y * 16;
    const size_t bOff = (size_t)blockIdx.z * K * HW;
    const size_t oOff = (size_t)blockIdx.z * M * HW;
    const unsigned ldsA = (unsigned)(size_t)(void*)&At[0][0];  // LDS aperture: low 32 bits
    const unsigned ldsB = (unsigned)(size_t)(void*)&Bt[0][0];
    v8f acc = {};
    const int ksteps = K >> 5;
    for (int ks = 0; ks < ksteps; ++ks) {
        // wave 0: DMA the 128x32 A tile; wave 1: DMA the 32x16 B tile (TENSORcnt)
        if (w == 0) {
            tdm_load_2d_b16(ldsA, A + (size_t)m0 * K + ks * 32, 32u, 128u, (unsigned long long)K);
            __builtin_amdgcn_s_wait_tensorcnt(0);
        } else if (w == 1) {
            tdm_load_2d_b16(ldsB, B + bOff + (size_t)(ks * 32) * HW + n0, 16u, 32u,
                            (unsigned long long)HW);
            __builtin_amdgcn_s_wait_tensorcnt(0);
        }
        __syncthreads();
        Frag af, bf;
        {
            int mr   = w * 16 + (lane & 15);
            int off4 = (lane < 16) ? 0 : 4;
            #pragma unroll
            for (int p = 0; p < 8; ++p) {
                int ku = ((p < 4) ? p : p + 4) + off4;  // u32 index = k0/2
                af.u[p] = At[mr][ku];
                bf.s[2 * p]     = ((const unsigned short*)&Bt[0][0])[(2 * ku) * 16 + (lane & 15)];
                bf.s[2 * p + 1] = ((const unsigned short*)&Bt[0][0])[(2 * ku + 1) * 16 + (lane & 15)];
            }
        }
        acc = wmma_bf16(af.v, bf.v, acc);
        __syncthreads();
    }
    const float RS = rsqrtf(1.0f + 1e-5f);
    int hb = lane >> 4, cl = lane & 15;
    #pragma unroll
    for (int r = 0; r < 8; ++r) {
        int row = m0 + w * 16 + r + 8 * hb;
        int col = n0 + cl;
        float v = gamma[row] * RS * acc[r] + beta[row];
        size_t oi = oOff + (size_t)row * HW + col;
        if (MODE == 0)      ((float*)outp)[oi]          = v;
        else if (MODE == 1) ((unsigned short*)outp)[oi] = f2bf(v);
        else                ((unsigned short*)outp)[oi] = f2bf(fmaxf(v, 0.f));
    }
}

// ---------------- fused relative-position attention ----------------
// one wave per (i-tile, g, b): flash-style online softmax over 784 j
__global__ __launch_bounds__(32) void attn_kernel(const unsigned short* __restrict__ qkv,
                                                  const float* __restrict__ rel,
                                                  const float* __restrict__ g_sim,
                                                  const float* __restrict__ b_sim,
                                                  const float* __restrict__ g_out,
                                                  const float* __restrict__ b_out,
                                                  float* __restrict__ binary) {
    const int lane = threadIdx.x;
    const int i0 = blockIdx.x * 16;
    const int g  = blockIdx.y;
    const int b  = blockIdx.z;
    const size_t base = ((size_t)b * OC2 + (size_t)g * 128) * HW;  // q:+0 k:+32*HW v:+64*HW
    const float RS  = rsqrtf(1.0f + 1e-5f);
    const float ss0 = g_sim[g] * RS,      sb0 = b_sim[g];
    const float ss1 = g_sim[8 + g] * RS,  sb1 = b_sim[8 + g];
    const float ss2 = g_sim[16 + g] * RS, sb2 = b_sim[16 + g];

    __shared__ float qld[32][16];
    __shared__ float kld[32][32];
    __shared__ float relq[32][48];
    __shared__ float relk[32][48];
    __shared__ float relv[64][48];
    __shared__ float P[16][32];
    __shared__ unsigned short PB[16][32];
    __shared__ float rowAlpha[16];
    __shared__ float rowL[16];
    __shared__ float outSV[64][16];
    __shared__ float outSVE[64][16];

    // stage q tile (c x i) as f32 for the qr VALU pass
    for (int e = lane; e < 32 * 16; e += 32) {
        int c = e >> 4, ii = e & 15;
        qld[c][ii] = bf2f(qkv[base + (size_t)c * HW + i0 + ii]);
    }
    // A-fragment of q^T (m=i rows, k=c)
    Frag aq;
    {
        int mr  = i0 + (lane & 15);
        int off = (lane < 16) ? 0 : 8;
        #pragma unroll
        for (int p = 0; p < 8; ++p) {
            int k0 = ((p < 4) ? 2 * p : 2 * p + 8) + off;
            aq.s[2 * p]     = qkv[base + (size_t)k0 * HW + mr];
            aq.s[2 * p + 1] = qkv[base + (size_t)(k0 + 1) * HW + mr];
        }
    }
    float m_i = -1e30f, l_i = 0.f;
    v8f accv[4];
    #pragma unroll
    for (int nt = 0; nt < 4; ++nt) { v8f z = {}; accv[nt] = z; }
    float sve[32];
    #pragma unroll
    for (int c = 0; c < 32; ++c) sve[c] = 0.f;

    const int hb = lane >> 4;

    for (int jt = 0; jt < 25; ++jt) {
        const int j0 = jt * 32;
        const int jw = (HW - j0 >= 32) ? 32 : (HW - j0);  // 16 on last tile
        // k tile (c x j) f32 for kr
        for (int c = 0; c < 32; ++c)
            kld[c][lane] = (lane < jw) ? bf2f(qkv[base + (size_t)(QK + c) * HW + j0 + lane]) : 0.f;
        // relative-embedding diagonal bands
        {
            const int dq0 = i0 - j0 + 752;  // + t(=row-col+31) gives d = i-j+783
            const int dk0 = j0 - i0 + 768;  // + t(=col-row+15) gives d = j-i+783
            for (int e = lane; e < 32 * 48; e += 32) {
                int c = e / 48, t = e % 48;
                relq[c][t] = rel[(size_t)c * RELW + iclamp(dq0 + t, 0, RELW - 1)];
                relk[c][t] = rel[(size_t)(QK + c) * RELW + iclamp(dk0 + t, 0, RELW - 1)];
            }
            for (int e = lane; e < 64 * 48; e += 32) {
                int c = e / 48, t = e % 48;
                relv[c][t] = rel[(size_t)(2 * QK + c) * RELW + iclamp(dq0 + t, 0, RELW - 1)];
            }
        }
        // qk via WMMA: B fragment = k tile (k=c per lane, n=j)
        Frag bk0, bk1;
        {
            const int kn = lane & 15, koff = (lane < 16) ? 0 : 8;
            #pragma unroll
            for (int p = 0; p < 8; ++p) {
                int k0 = ((p < 4) ? 2 * p : 2 * p + 8) + koff;
                const unsigned short* kp = qkv + base + (size_t)(QK + k0) * HW + j0 + kn;
                bk0.s[2 * p]     = kp[0];
                bk0.s[2 * p + 1] = kp[HW];
                if (jw == 32) {
                    bk1.s[2 * p]     = kp[16];
                    bk1.s[2 * p + 1] = kp[HW + 16];
                }
            }
        }
        v8f z = {};
        v8f s0 = wmma_bf16(aq.v, bk0.v, z);
        v8f s1 = z;
        if (jw == 32) s1 = wmma_bf16(aq.v, bk1.v, z);
        __syncthreads();
        // scatter scores to LDS (D layout: n=lane%16, m=r+8*half)
        {
            int col = lane & 15;
            #pragma unroll
            for (int r = 0; r < 8; ++r) {
                int row = r + 8 * hb;
                P[row][col]      = s0[r];
                P[row][col + 16] = (jw == 32) ? s1[r] : -1e30f;
            }
        }
        __syncthreads();
        // add qr/kr (VALU) + sim BN
        for (int e = lane; e < 512; e += 32) {
            int row = e >> 5, col = e & 31;
            float v;
            if (col < jw) {
                float qr = 0.f, kr = 0.f;
                #pragma unroll
                for (int c = 0; c < 32; ++c) {
                    qr = fmaf(qld[c][row], relq[c][row - col + 31], qr);
                    kr = fmaf(kld[c][col], relk[c][col - row + 15], kr);
                }
                v = ss0 * P[row][col] + sb0 + ss1 * qr + sb1 + ss2 * kr + sb2;
            } else v = -1e30f;
            P[row][col] = v;
        }
        __syncthreads();
        // online softmax: lane pair (L, L+16) owns row L%16; halves split columns
        float alpha;
        {
            int row = lane & 15;
            float mt = -1e30f;
            #pragma unroll
            for (int col = 0; col < 32; ++col) mt = fmaxf(mt, P[row][col]);
            float m_new = fmaxf(m_i, mt);
            alpha = __expf(m_i - m_new);
            float sh = 0.f;
            #pragma unroll
            for (int t = 0; t < 16; ++t) {
                int col = 16 * hb + t;
                float pv = P[row][col];
                float p  = (pv > -1e29f) ? __expf(pv - m_new) : 0.f;
                P[row][col]  = p;
                PB[row][col] = f2bf(p);
                sh += p;
            }
            float s = sh + __shfl_xor(sh, 16);
            l_i = alpha * l_i + s;
            m_i = m_new;
            if (lane < 16) rowAlpha[row] = alpha;
        }
        __syncthreads();
        // sve accumulate (VALU): lane owns i=lane%16, half splits c
        {
            int i_loc  = lane & 15;
            int cbase  = hb * 32;
            #pragma unroll
            for (int c = 0; c < 32; ++c) {
                float a = sve[c] * alpha;
                const float* rv = &relv[cbase + c][0];
                #pragma unroll 8
                for (int col = 0; col < 32; ++col)
                    a = fmaf(P[i_loc][col], rv[i_loc - col + 31], a);
                sve[c] = a;
            }
        }
        // P @ V^T via WMMA (rescale accumulators first)
        {
            #pragma unroll
            for (int nt = 0; nt < 4; ++nt)
                #pragma unroll
                for (int r = 0; r < 8; ++r) accv[nt][r] *= rowAlpha[r + 8 * hb];
            Frag ap;
            {
                int row = lane & 15;
                int off = (lane < 16) ? 0 : 8;
                #pragma unroll
                for (int p = 0; p < 8; ++p) {
                    int k0 = ((p < 4) ? 2 * p : 2 * p + 8) + off;
                    ap.s[2 * p]     = PB[row][k0];
                    ap.s[2 * p + 1] = PB[row][k0 + 1];
                }
            }
            const int vn = lane & 15, voff = (lane < 16) ? 0 : 8;
            const size_t vrow = base + (size_t)(2 * QK) * HW + j0;
            #pragma unroll
            for (int nt = 0; nt < 4; ++nt) {
                Frag bv;
                #pragma unroll
                for (int p = 0; p < 8; ++p) {
                    int k0 = ((p < 4) ? 2 * p : 2 * p + 8) + voff;
                    size_t ci = vrow + (size_t)(nt * 16 + vn) * HW + k0;
                    bv.s[2 * p]     = (k0 < jw) ? qkv[ci] : (unsigned short)0;
                    bv.s[2 * p + 1] = (k0 + 1 < jw) ? qkv[ci + 1] : (unsigned short)0;
                }
                accv[nt] = wmma_bf16(ap.v, bv.v, accv[nt]);
            }
        }
        __syncthreads();
    }
    if (lane < 16) rowL[lane] = l_i;
    __syncthreads();
    // normalize + transpose to (c, i) in LDS
    {
        int cl = lane & 15;
        #pragma unroll
        for (int nt = 0; nt < 4; ++nt)
            #pragma unroll
            for (int r = 0; r < 8; ++r) {
                int row = r + 8 * hb;
                outSV[nt * 16 + cl][row] = accv[nt][r] / rowL[row];
            }
        int i_loc = lane & 15;
        int cbase = hb * 32;
        float inv = 1.f / l_i;
        #pragma unroll
        for (int c = 0; c < 32; ++c) outSVE[cbase + c][i_loc] = sve[c] * inv;
    }
    __syncthreads();
    // BN (g_out over 1024 interleaved channels) + pair-sum -> binary
    for (int e = lane; e < 64 * 16; e += 32) {
        int c = e >> 4, ii = e & 15;
        int ch = g * 128 + 2 * c;
        float val = g_out[ch] * RS * outSV[c][ii] + b_out[ch]
                  + g_out[ch + 1] * RS * outSVE[c][ii] + b_out[ch + 1];
        binary[((size_t)b * BR + (size_t)g * 64 + c) * HW + i0 + ii] = val;
    }
}

// ---------------- gate input: relu(concat(unary, binary)) -> bf16 ----------------
__global__ __launch_bounds__(256) void gin_kernel(const float* __restrict__ unary,
                                                  const float* __restrict__ binary,
                                                  unsigned short* __restrict__ gin) {
    int e = blockIdx.x * 256 + threadIdx.x;
    if (e >= NB * OC2 * HW) return;
    int b = e / (OC2 * HW);
    int rem = e % (OC2 * HW);
    int ch = rem / HW;
    int i  = rem % HW;
    float v = (ch < BR) ? unary[((size_t)b * BR + ch) * HW + i]
                        : binary[((size_t)b * BR + (ch - BR)) * HW + i];
    gin[e] = f2bf(fmaxf(v, 0.f));
}

// ---------------- final: sigmoid(g)*binary + unary ----------------
__global__ __launch_bounds__(256) void final_kernel(const float* __restrict__ gbuf,
                                                    const float* __restrict__ binary,
                                                    const float* __restrict__ unary,
                                                    float* __restrict__ out) {
    int e = blockIdx.x * 256 + threadIdx.x;
    if (e >= NB * BR * HW) return;
    float gate = 1.f / (1.f + __expf(-gbuf[e]));
    out[e] = gate * binary[e] + unary[e];
}

// ---------------- host side ----------------
extern "C" void kernel_launch(void* const* d_in, const int* in_sizes, int n_in,
                              void* d_out, int out_size, void* d_ws, size_t ws_size,
                              hipStream_t stream) {
    const float* x     = (const float*)d_in[0];
    const float* W_qkv = (const float*)d_in[1];
    const float* g_qkv = (const float*)d_in[2];
    const float* b_qkv = (const float*)d_in[3];
    const float* rel   = (const float*)d_in[4];
    const float* g_sim = (const float*)d_in[5];
    const float* b_sim = (const float*)d_in[6];
    const float* g_out = (const float*)d_in[7];
    const float* b_out = (const float*)d_in[8];
    const float* W_x3  = (const float*)d_in[9];
    const float* W_x1  = (const float*)d_in[10];
    const float* g_x   = (const float*)d_in[11];
    const float* b_x   = (const float*)d_in[12];
    const float* W_r   = (const float*)d_in[13];
    const float* g_r   = (const float*)d_in[14];
    const float* b_r   = (const float*)d_in[15];
    const float* W_p   = (const float*)d_in[16];
    const float* g_p   = (const float*)d_in[17];
    const float* b_p   = (const float*)d_in[18];

    char* ws = (char*)d_ws;
    size_t off = 0;
    auto alloc = [&](size_t bytes) { char* p = ws + off; off += (bytes + 255) & ~(size_t)255; return p; };
    unsigned short* xbf   = (unsigned short*)alloc((size_t)NB * CIN * HW * 2);
    unsigned short* wqkv  = (unsigned short*)alloc((size_t)OC2 * CIN * 2);
    unsigned short* qkvbf = (unsigned short*)alloc((size_t)NB * OC2 * HW * 2);
    unsigned short* ubf   = (unsigned short*)alloc((size_t)NB * BR * HW * 2);
    float*          unary = (float*)alloc((size_t)NB * BR * HW * 4);
    float*          binry = (float*)alloc((size_t)NB * BR * HW * 4);
    unsigned short* wx1   = (unsigned short*)alloc((size_t)BR * BR * 2);
    unsigned short* wr    = (unsigned short*)alloc((size_t)BR * OC2 * 2);
    unsigned short* wp    = (unsigned short*)alloc((size_t)BR * BR * 2);
    unsigned short* gin   = (unsigned short*)alloc((size_t)NB * OC2 * HW * 2);
    unsigned short* rbf   = (unsigned short*)alloc((size_t)NB * BR * HW * 2);
    float*          gbuf  = (float*)alloc((size_t)NB * BR * HW * 4);

    auto blocks = [](int n) { return (n + 255) / 256; };

    // bf16 packs
    cvt_kernel<<<blocks(NB * CIN * HW), 256, 0, stream>>>(x, xbf, NB * CIN * HW);
    cvt_kernel<<<blocks(OC2 * CIN), 256, 0, stream>>>(W_qkv, wqkv, OC2 * CIN);
    cvt_kernel<<<blocks(BR * BR), 256, 0, stream>>>(W_x1, wx1, BR * BR);
    cvt_kernel<<<blocks(BR * OC2), 256, 0, stream>>>(W_r, wr, BR * OC2);
    cvt_kernel<<<blocks(BR * BR), 256, 0, stream>>>(W_p, wp, BR * BR);

    // unary path: grouped 3x3 conv -> bf16, then 1x1 WMMA GEMM with BN
    conv3_kernel<<<blocks(NB * BR * HW), 256, 0, stream>>>(x, W_x3, ubf);

    // qkv projection: (1024x512) @ (512x784) x4 batches, BN fused, bf16 out
    gemm_bf16_kernel<1><<<dim3(OC2 / 128, HW / 16, NB), 256, 0, stream>>>(
        wqkv, xbf, g_qkv, b_qkv, qkvbf, OC2, CIN);

    gemm_bf16_kernel<0><<<dim3(BR / 128, HW / 16, NB), 256, 0, stream>>>(
        wx1, ubf, g_x, b_x, unary, BR, BR);

    // fused attention -> binary
    attn_kernel<<<dim3(HW / 16, G, NB), 32, 0, stream>>>(
        qkvbf, rel, g_sim, b_sim, g_out, b_out, binry);

    // gate path
    gin_kernel<<<blocks(NB * OC2 * HW), 256, 0, stream>>>(unary, binry, gin);

    gemm_bf16_kernel<2><<<dim3(BR / 128, HW / 16, NB), 256, 0, stream>>>(
        wr, gin, g_r, b_r, rbf, BR, OC2);

    gemm_bf16_kernel<0><<<dim3(BR / 128, HW / 16, NB), 256, 0, stream>>>(
        wp, rbf, g_p, b_p, gbuf, BR, BR);

    final_kernel<<<blocks(NB * BR * HW), 256, 0, stream>>>(gbuf, binry, unary, (float*)d_out);
}
